// TriangleMultiplication_16011638079849
// MI455X (gfx1250) — compile-verified, matrix operand order checked by reading, not run on hardware
//
#include <hip/hip_runtime.h>
#include <hip/hip_bf16.h>

// ---------------------------------------------------------------------------
// TriangleMultiplication (outgoing), B=1, N=512, D=H=128, fp32 in/out.
// bf16 WMMA (v_wmma_f32_16x16x32_bf16) for every contraction; intermediates
// a,b stored bf16 in [H,N,N] layout so the triangle einsum is 128 NT GEMMs.
// Weights pre-converted to bf16 once (k_conv). k_tri uses async global->LDS
// staging (ASYNCcnt) when the toolchain exposes the builtin.
// Workspace: a_t 64MB + b_t 64MB + x_t(f32) 128MB + weights 192KB.
// ---------------------------------------------------------------------------

#define NN 512
#define DD 128
#define HH 128
#define PLANE (512L * 512L)   // elements per h-plane
#define EPS 1e-5f
#define MASK_SCALE 0.04419417382415922f   // 512^-0.5

#if defined(__has_builtin)
#  if __has_builtin(__builtin_amdgcn_global_load_async_to_lds_b128)
#    define HAVE_ASYNC_LDS 1
#  endif
#  if __has_builtin(__builtin_amdgcn_s_wait_asynccnt)
#    define HAVE_WAIT_ASYNC_BUILTIN 1
#  endif
#endif

typedef __attribute__((ext_vector_type(16))) __bf16 v16bf;
typedef __attribute__((ext_vector_type(8)))  float  v8f;
typedef int i32x4 __attribute__((vector_size(4 * sizeof(int))));

union BF16x16 {
    v16bf v;
    unsigned short u[16];
    uint4 q[2];
};

union Pack8 {
    uint4 q;
    unsigned short u[8];
};

// native single-instruction fp32 -> bf16 (RNE) convert
static __device__ inline unsigned short f2bf(float f) {
    return __builtin_bit_cast(unsigned short, (__bf16)f);
}

// 16 contiguous bf16 values (LDS or global) -> WMMA operand row for one lane
static __device__ inline v16bf row16(const unsigned short* p) {
    BF16x16 r;
    r.q[0] = *(const uint4*)(p);
    r.q[1] = *(const uint4*)(p + 8);
    return r.v;
}

static __device__ inline v8f wmma_bf16(v16bf a, v16bf b, v8f c) {
    return __builtin_amdgcn_wmma_f32_16x16x32_bf16(
        /*neg_a=*/false, a, /*neg_b=*/false, b,
        /*c_mod=*/(short)0, c, /*reuse_a=*/false, /*reuse_b=*/false);
}

// ---------------------------------------------------------------------------
// Kernel 0: one-shot weight conversion fp32 -> bf16 into workspace.
// Layout: [w_g 16384][w_ab_p 32768][w_ab_g 32768][w_z 16384] = 98304 elems.
// ---------------------------------------------------------------------------
__global__ __launch_bounds__(256) void k_conv(
    const float* __restrict__ w_g, const float* __restrict__ w_ab_p,
    const float* __restrict__ w_ab_g, const float* __restrict__ w_z,
    unsigned short* __restrict__ out)
{
    const int idx = (blockIdx.x * 256 + threadIdx.x) * 4;   // < 98304
    const float* src;
    int off;
    if (idx < 16384)      { src = w_g;    off = idx; }
    else if (idx < 49152) { src = w_ab_p; off = idx - 16384; }
    else if (idx < 81920) { src = w_ab_g; off = idx - 49152; }
    else                  { src = w_z;    off = idx - 81920; }
    float4 v = *(const float4*)(src + off);
    unsigned int lo = (unsigned int)f2bf(v.x) | ((unsigned int)f2bf(v.y) << 16);
    unsigned int hi = (unsigned int)f2bf(v.z) | ((unsigned int)f2bf(v.w) << 16);
    uint2 pk; pk.x = lo; pk.y = hi;
    *(uint2*)(out + idx) = pk;
}

// ---------------------------------------------------------------------------
// Kernel 1: input LayerNorm + projections (g, p, gate) + gating epilogue.
// Block = 256 threads (8 waves), handles 16 flattened (i,j) rows.
// Wave w: g column-tile w (cols 16w..16w+15) and h-tiles {2w, 2w+1} of p/gate.
// ---------------------------------------------------------------------------
__global__ __launch_bounds__(256) void k_proj(
    const float* __restrict__ z, const float* __restrict__ mask,
    const float* __restrict__ ln_w, const float* __restrict__ ln_b,
    const unsigned short* __restrict__ wg_bf,   // [128][128] bf16
    const unsigned short* __restrict__ wp_bf,   // [256][128] bf16
    const unsigned short* __restrict__ wt_bf,   // [256][128] bf16
    const float* __restrict__ b_ab_p, const float* __restrict__ b_ab_g,
    unsigned short* __restrict__ a_t, unsigned short* __restrict__ b_t,
    float* __restrict__ g_out)
{
    __shared__ float zf[16][DD];
    __shared__ float red[16][8];
    __shared__ float mean_s[16], rstd_s[16], ms[16];
    __shared__ __align__(16) unsigned short zs[16][DD];

    const int  tid  = threadIdx.x;
    const long row0 = (long)blockIdx.x * 16;

    // --- load 16 rows of z (fp32) ---
    {
        const int base = tid * 8;          // 0..2047
        const int r = base >> 7, c = base & 127;
        const float* src = z + row0 * DD + base;
        float4 v0 = *(const float4*)(src);
        float4 v1 = *(const float4*)(src + 4);
        *(float4*)&zf[r][c]     = v0;
        *(float4*)&zf[r][c + 4] = v1;
    }
    __syncthreads();

    // --- mean ---
    if (tid < 128) {
        const int r = tid >> 3, p = tid & 7;
        float s = 0.f;
#pragma unroll
        for (int e = 0; e < 16; ++e) s += zf[r][p * 16 + e];
        red[r][p] = s;
    }
    __syncthreads();
    if (tid < 16) {
        float s = 0.f;
#pragma unroll
        for (int p = 0; p < 8; ++p) s += red[tid][p];
        mean_s[tid] = s * (1.f / 128.f);
    }
    __syncthreads();
    // --- variance ---
    if (tid < 128) {
        const int r = tid >> 3, p = tid & 7;
        const float m = mean_s[r];
        float s = 0.f;
#pragma unroll
        for (int e = 0; e < 16; ++e) {
            float d = zf[r][p * 16 + e] - m;
            s += d * d;
        }
        red[r][p] = s;
    }
    __syncthreads();
    if (tid < 16) {
        float s = 0.f;
#pragma unroll
        for (int p = 0; p < 8; ++p) s += red[tid][p];
        rstd_s[tid] = rsqrtf(s * (1.f / 128.f) + EPS);
        ms[tid] = mask[row0 + tid] * MASK_SCALE;
    }
    __syncthreads();
    // --- normalize -> bf16 ---
    {
        const int base = tid * 8;
        const int r = base >> 7, c = base & 127;
        const float m = mean_s[r], rs = rstd_s[r];
#pragma unroll
        for (int e = 0; e < 8; ++e)
            zs[r][c + e] = f2bf((zf[r][c + e] - m) * rs * ln_w[c + e] + ln_b[c + e]);
    }
    __syncthreads();

    // --- WMMA projections ---
    const int w    = tid >> 5;
    const int lane = tid & 31;
    const int lr   = lane & 15;            // M (A) / N (B) within tile
    const int lk   = (lane >> 4) * 16;     // K half select
    const int rh   = (lane >> 4) * 8;      // C/D row half

    // g tile: cols 16w .. 16w+15
    {
        v8f acc = {};
#pragma unroll
        for (int k0 = 0; k0 < DD; k0 += 32) {
            v16bf av = row16(&zs[lr][k0 + lk]);
            v16bf bv = row16(wg_bf + (w * 16 + lr) * DD + k0 + lk);
            acc = wmma_bf16(av, bv, acc);
        }
        const int c = w * 16 + lr;
#pragma unroll
        for (int r = 0; r < 8; ++r)
            g_out[(row0 + rh + r) * DD + c] = acc[r];
    }

    // p/gate tiles: h-tiles 2w and 2w+1 (h columns of the 256-wide ab)
#pragma unroll
    for (int tt = 0; tt < 2; ++tt) {
        const int t    = w * 2 + tt;       // 0..15
        const int hcol = t * 16 + lr;      // 0..255
        v8f accP = {};
        v8f accG = {};
#pragma unroll
        for (int k0 = 0; k0 < DD; k0 += 32) {
            v16bf av = row16(&zs[lr][k0 + lk]);
            v16bf bp = row16(wp_bf + hcol * DD + k0 + lk);
            accP = wmma_bf16(av, bp, accP);
            v16bf bg = row16(wt_bf + hcol * DD + k0 + lk);
            accG = wmma_bf16(av, bg, accG);
        }
        const float bp0 = b_ab_p[hcol];
        const float bg0 = b_ab_g[hcol];
        unsigned short* dst = (hcol < HH)
            ? (a_t + (long)hcol * PLANE)
            : (b_t + (long)(hcol - HH) * PLANE);
        Pack8 pk;
#pragma unroll
        for (int r = 0; r < 8; ++r) {
            const int rr = rh + r;
            float gate = 1.f / (1.f + __expf(-(accG[r] + bg0)));
            float val  = (accP[r] + bp0) * ms[rr] * gate;
            pk.u[r] = f2bf(val);
        }
        // rows rh..rh+7 are contiguous in the [h][row] plane -> one 16B store
        *(uint4*)(dst + row0 + rh) = pk.q;
    }
}

// ---------------------------------------------------------------------------
// Kernel 2: triangle einsum. For each h: C = A * B^T, 512x512x512 bf16->f32.
// Block = 256 threads, 128x64 C tile, 2x2 wmma tiles per wave (A/B fragment
// reuse: 4 wmma per 4 operand fetches). LDS: A 128x32 + B 64x32 bf16 = 12KB.
// Staging uses GLOBAL_LOAD_ASYNC_TO_LDS_B128 (ASYNCcnt) when available.
// grid = (32 C-tiles, 128 h)
// ---------------------------------------------------------------------------
__global__ __launch_bounds__(256) void k_tri(
    const unsigned short* __restrict__ a_t,
    const unsigned short* __restrict__ b_t,
    float* __restrict__ x_t)
{
    __shared__ __align__(16) unsigned short As[128][32];
    __shared__ __align__(16) unsigned short Bs[64][32];

    const int tid = threadIdx.x;
    const int h   = blockIdx.y;
    const int i0  = (blockIdx.x >> 3) * 128;   // 4 row blocks
    const int j0  = (blockIdx.x & 7) * 64;     // 8 col blocks
    const long plane = (long)h * PLANE;
    const unsigned short* A = a_t + plane;
    const unsigned short* B = b_t + plane;

    const int w    = tid >> 5;
    const int lane = tid & 31;
    const int lr   = lane & 15;
    const int lk   = (lane >> 4) * 16;
    const int rh   = (lane >> 4) * 8;
    const int tra  = (w >> 1) * 16;            // A row tiles: tra, tra+64
    const int tca  = (w & 1) * 16;             // B col tiles: tca, tca+32

    // staging slots: A = 512 chunks of 8 ushorts, 2/thread; B = 256, 1/thread
    const int ca  = tid * 2;
    const int ar  = ca >> 2, ac = (ca & 3) * 8;
    const int br  = tid >> 2, bc = (tid & 3) * 8;

    v8f acc00 = {}, acc01 = {}, acc10 = {}, acc11 = {};

    for (int k0 = 0; k0 < NN; k0 += 32) {
        const unsigned short* gA = A + (long)(i0 + ar) * NN + k0 + ac;
        const unsigned short* gB = B + (long)(j0 + br) * NN + k0 + bc;
#if defined(HAVE_ASYNC_LDS)
        __builtin_amdgcn_global_load_async_to_lds_b128(
            (i32x4*)gA,       (i32x4*)&As[ar][ac],     0, 0);
        __builtin_amdgcn_global_load_async_to_lds_b128(
            (i32x4*)(gA + 8), (i32x4*)&As[ar][ac + 8], 0, 0);
        __builtin_amdgcn_global_load_async_to_lds_b128(
            (i32x4*)gB,       (i32x4*)&Bs[br][bc],     0, 0);
#  if defined(HAVE_WAIT_ASYNC_BUILTIN)
        __builtin_amdgcn_s_wait_asynccnt(0);
#  else
        asm volatile("s_wait_asynccnt 0x0" ::: "memory");
#  endif
#else
        *(uint4*)&As[ar][ac]     = *(const uint4*)(gA);
        *(uint4*)&As[ar][ac + 8] = *(const uint4*)(gA + 8);
        *(uint4*)&Bs[br][bc]     = *(const uint4*)(gB);
#endif
        if (k0 + 32 < NN) {   // warm caches for next panel (global_prefetch_b8)
            __builtin_prefetch(gA + 32, 0, 3);
            __builtin_prefetch(gB + 32, 0, 3);
        }
        __syncthreads();
        v16bf a0 = row16(&As[tra + lr][lk]);
        v16bf a1 = row16(&As[tra + 64 + lr][lk]);
        v16bf b0 = row16(&Bs[tca + lr][lk]);
        v16bf b1 = row16(&Bs[tca + 32 + lr][lk]);
        acc00 = wmma_bf16(a0, b0, acc00);
        acc01 = wmma_bf16(a0, b1, acc01);
        acc10 = wmma_bf16(a1, b0, acc10);
        acc11 = wmma_bf16(a1, b1, acc11);
        __syncthreads();
    }

    float* C = x_t + plane;
    const int cj0 = j0 + tca + lr;
    const int cj1 = cj0 + 32;
#pragma unroll
    for (int r = 0; r < 8; ++r) {
        const long ri0 = (long)(i0 + tra + rh + r) * NN;
        const long ri1 = (long)(i0 + tra + 64 + rh + r) * NN;
        C[ri0 + cj0] = acc00[r];
        C[ri0 + cj1] = acc01[r];
        C[ri1 + cj0] = acc10[r];
        C[ri1 + cj1] = acc11[r];
    }
}

// ---------------------------------------------------------------------------
// Kernel 3: gather x[:,row], LayerNorm over H, project with w_z -> out x.
// Block = 256 threads, 16 rows.
// ---------------------------------------------------------------------------
__global__ __launch_bounds__(256) void k_out(
    const float* __restrict__ x_t,
    const float* __restrict__ ln_w, const float* __restrict__ ln_b,
    const unsigned short* __restrict__ wz_bf,   // [128][128] bf16
    float* __restrict__ out_x)
{
    __shared__ float xs[16][HH];
    __shared__ float red[16][8];
    __shared__ float mean_s[16], rstd_s[16];
    __shared__ __align__(16) unsigned short ys[16][HH];

    const int  tid  = threadIdx.x;
    const long row0 = (long)blockIdx.x * 16;

    // gather-transpose: thread loads 8 consecutive rows of one h-plane
    {
        const int hh = tid >> 1;
        const int rb = (tid & 1) * 8;
        const float* src = x_t + (long)hh * PLANE + row0 + rb;
        float4 v0 = *(const float4*)(src);
        float4 v1 = *(const float4*)(src + 4);
        xs[rb + 0][hh] = v0.x; xs[rb + 1][hh] = v0.y;
        xs[rb + 2][hh] = v0.z; xs[rb + 3][hh] = v0.w;
        xs[rb + 4][hh] = v1.x; xs[rb + 5][hh] = v1.y;
        xs[rb + 6][hh] = v1.z; xs[rb + 7][hh] = v1.w;
    }
    __syncthreads();

    if (tid < 128) {
        const int r = tid >> 3, p = tid & 7;
        float s = 0.f;
#pragma unroll
        for (int e = 0; e < 16; ++e) s += xs[r][p * 16 + e];
        red[r][p] = s;
    }
    __syncthreads();
    if (tid < 16) {
        float s = 0.f;
#pragma unroll
        for (int p = 0; p < 8; ++p) s += red[tid][p];
        mean_s[tid] = s * (1.f / 128.f);
    }
    __syncthreads();
    if (tid < 128) {
        const int r = tid >> 3, p = tid & 7;
        const float m = mean_s[r];
        float s = 0.f;
#pragma unroll
        for (int e = 0; e < 16; ++e) {
            float d = xs[r][p * 16 + e] - m;
            s += d * d;
        }
        red[r][p] = s;
    }
    __syncthreads();
    if (tid < 16) {
        float s = 0.f;
#pragma unroll
        for (int p = 0; p < 8; ++p) s += red[tid][p];
        rstd_s[tid] = rsqrtf(s * (1.f / 128.f) + EPS);
    }
    __syncthreads();
    {
        const int base = tid * 8;
        const int r = base >> 7, c = base & 127;
        const float m = mean_s[r], rs = rstd_s[r];
#pragma unroll
        for (int e = 0; e < 8; ++e)
            ys[r][c + e] = f2bf((xs[r][c + e] - m) * rs * ln_w[c + e] + ln_b[c + e]);
    }
    __syncthreads();

    const int w    = tid >> 5;
    const int lane = tid & 31;
    const int lr   = lane & 15;
    const int lk   = (lane >> 4) * 16;
    const int rh   = (lane >> 4) * 8;

    v8f acc = {};
#pragma unroll
    for (int k0 = 0; k0 < HH; k0 += 32) {
        v16bf av = row16(&ys[lr][k0 + lk]);
        v16bf bv = row16(wz_bf + (w * 16 + lr) * HH + k0 + lk);
        acc = wmma_bf16(av, bv, acc);
    }
    const int cd = w * 16 + lr;
#pragma unroll
    for (int r = 0; r < 8; ++r)
        out_x[(row0 + rh + r) * DD + cd] = acc[r];
}

extern "C" void kernel_launch(void* const* d_in, const int* in_sizes, int n_in,
                              void* d_out, int out_size, void* d_ws, size_t ws_size,
                              hipStream_t stream)
{
    const float* z       = (const float*)d_in[0];
    const float* mask    = (const float*)d_in[1];
    const float* ln_in_w = (const float*)d_in[2];
    const float* ln_in_b = (const float*)d_in[3];
    const float* w_ab_p  = (const float*)d_in[4];
    const float* b_ab_p  = (const float*)d_in[5];
    const float* w_ab_g  = (const float*)d_in[6];
    const float* b_ab_g  = (const float*)d_in[7];
    const float* w_g     = (const float*)d_in[8];
    const float* ln_o_w  = (const float*)d_in[9];
    const float* ln_o_b  = (const float*)d_in[10];
    const float* w_z     = (const float*)d_in[11];

    float* out_x = (float*)d_out;                       // x: [1,512,512,128]
    float* out_g = out_x + (size_t)NN * NN * DD;        // g: [1,512,512,128]

    unsigned short* a_t = (unsigned short*)d_ws;        // [128][512][512] bf16
    unsigned short* b_t = a_t + (size_t)HH * PLANE;     // [128][512][512] bf16
    float*          x_t = (float*)(b_t + (size_t)HH * PLANE); // [128][512][512] f32
    unsigned short* wbf = (unsigned short*)(x_t + (size_t)HH * PLANE);
    unsigned short* wg_bf = wbf;            // 16384
    unsigned short* wp_bf = wg_bf + 16384;  // 32768
    unsigned short* wt_bf = wp_bf + 32768;  // 32768
    unsigned short* wz_bf = wt_bf + 32768;  // 16384

    const int row_blocks = (NN * NN) / 16;              // 16384

    k_conv<<<96, 256, 0, stream>>>(w_g, w_ab_p, w_ab_g, w_z, wbf);
    k_proj<<<row_blocks, 256, 0, stream>>>(z, mask, ln_in_w, ln_in_b,
                                           wg_bf, wp_bf, wt_bf,
                                           b_ab_p, b_ab_g,
                                           a_t, b_t, out_g);
    k_tri<<<dim3(32, HH), 256, 0, stream>>>(a_t, b_t, x_t);
    k_out<<<row_blocks, 256, 0, stream>>>(x_t, ln_o_w, ln_o_b, wz_bf, out_x);
}
